// TemporalLayer_5342939316374
// MI455X (gfx1250) — matrix operand order repaired and checked by reading, not compiled, not run on hardware
//
#include <hip/hip_runtime.h>
#include <hip/hip_bf16.h>

// ---------------------------------------------------------------------------
// TemporalLayer on MI455X (gfx1250): bf16 WMMA pipeline.
//   B=2, T=256, P=64, E=512, NH=8, KQ=64, BP=B*P=128, H=NH*KQ=512
// All matmuls use v_wmma_f32_16x16x32_bf16 (f32 accumulate).
// Workspace layout (bf16 elements unless noted):
//   [0          .. 786432)   Wq/Wk/Wv packed as B-fragments
//   [786432     .. 1048576)  Wo packed as B-fragments
//   [1048576    .. +16.7M)   q as A-fragments   [bp][head][mt16][ks2][lane32][e16]
//   next 16.7M               k^T as B-fragments [bp][head][nts16][ks2][lane32][e16]
//   next 16.7M               v as B-fragments   [bp][head][ntd4][ks8][lane32][e16]
//   next 16.7M               attn-out as A-frag [bp][mt16][ks16][lane32][e16]
//   then float[1536]         bq|bk|bv concatenated (branch-free bias fetch)
// Total ~130 MB -> resident in the 192 MB global L2.
// ---------------------------------------------------------------------------

typedef __attribute__((ext_vector_type(16))) __bf16 v16bf;
typedef __attribute__((ext_vector_type(8)))  float  v8f;

__device__ __forceinline__ v8f wmma_bf16(v16bf a, v16bf b, v8f c) {
  // (neg_a, A, neg_b, B, c_mod, C, reuse_a, reuse_b)
  return __builtin_amdgcn_wmma_f32_16x16x32_bf16(false, a, false, b, (short)0, c,
                                                 false, false);
}

// ---------------------------------------------------------------------------
// K0a: pack the four 512x512 fp32 weight matrices into bf16 B-fragment layout.
// B (32x16, 16-bit): frag[e] = W[K = (e&7) + 8*(lane>=16) + 16*(e>=8)][col = lane&15]
// Linear layout: [m4][nt32][ks16][lane32][e16]
// ---------------------------------------------------------------------------
__global__ void pack_weights(const float* __restrict__ Wq,
                             const float* __restrict__ Wk,
                             const float* __restrict__ Wv,
                             const float* __restrict__ Wo,
                             __bf16* __restrict__ wfrag) {
  int idx = blockIdx.x * 256 + threadIdx.x;   // < 4*262144
  int m    = idx >> 18;
  int rem  = idx & 262143;
  int nt   = rem >> 13;
  int r2   = rem & 8191;
  int ks   = r2 >> 9;
  int r3   = r2 & 511;
  int lane = r3 >> 4;
  int e    = r3 & 15;
  int K    = (e & 7) + ((lane >> 4) << 3) + ((e >> 3) << 4);
  int col  = lane & 15;
  int row  = ks * 32 + K;         // contraction index (E or H)
  int n    = nt * 16 + col;       // output column
  const float* W = (m == 0) ? Wq : (m == 1) ? Wk : (m == 2) ? Wv : Wo;
  wfrag[idx] = (__bf16)W[row * 512 + n];
}

// K0b: concatenate the three projection biases -> branch-free fetch in K1.
__global__ void pack_bias(const float* __restrict__ bq,
                          const float* __restrict__ bk,
                          const float* __restrict__ bv,
                          float* __restrict__ biasCat) {
  int i = blockIdx.x * 256 + threadIdx.x;   // < 1536
  float v = (i < 512) ? bq[i] : (i < 1024) ? bk[i - 512] : bv[i - 1024];
  biasCat[i] = v;
}

// ---------------------------------------------------------------------------
// K1: fused QKV projection. Block = (bp, mt) covering 16 t-rows; 8 waves.
// x tile staged into LDS in A-fragment layout; each wave computes 12 16x16
// output tiles (96 across q|k|v), scatter-stores into attention frag layouts.
// ---------------------------------------------------------------------------
__global__ void __launch_bounds__(256)
qkv_kernel(const float* __restrict__ x,
           const float* __restrict__ biasCat,
           const __bf16* __restrict__ wfrag,
           __bf16* __restrict__ qf, __bf16* __restrict__ kf,
           __bf16* __restrict__ vf) {
  __shared__ __align__(32) __bf16 ldsA[16 * 32 * 16];   // [ks][lane][e], 16 KB
  int bp = blockIdx.x >> 4;
  int mt = blockIdx.x & 15;
  int b = bp >> 6, p = bp & 63;
  int tid = threadIdx.x;

  // --- stage x[bp, t0:t0+16, 0:512] (fp32) -> bf16 A-fragments in LDS ---
  {
    int row = tid >> 4;                 // 0..15 (t within tile)
    int e0  = (tid & 15) * 32;          // 32 consecutive E per thread
    int t   = mt * 16 + row;
    const float* xrow = x + (size_t)((b * 256 + t) * 64 + p) * 512;
    for (int i = 0; i < 32; i += 4) {
      float4 xv = *(const float4*)(xrow + e0 + i);
      float vals[4] = {xv.x, xv.y, xv.z, xv.w};
#pragma unroll
      for (int j = 0; j < 4; ++j) {
        int eg = e0 + i + j;
        int ks = eg >> 5, kk = eg & 31;
        int hi = (kk >> 3) & 1;
        int lp = row | (hi << 4);
        int ep = (kk & 7) + ((kk >> 4) << 3);
        ldsA[(ks * 32 + lp) * 16 + ep] = (__bf16)vals[j];
      }
    }
  }
  __syncthreads();

  // wave id is provably uniform -> scalar branches for the m=q/k/v paths
  int w    = __builtin_amdgcn_readfirstlane(tid >> 5);
  int lane = tid & 31, col = lane & 15;
  for (int it = 0; it < 12; ++it) {
    int ng = w + it * 8;            // 0..95 : [m=q/k/v][nt 0..31]  (scalar)
    int m  = ng >> 5;
    int nt = ng & 31;
    v8f acc = {};
    const __bf16* wb = wfrag + (size_t)((m * 32 + nt) * 16) * 512;
    // prefetch next wave-iteration's weight tile into L2/L0 (global_prefetch_b8)
    if (it + 1 < 12) {
      int ng2 = w + (it + 1) * 8;
      __builtin_prefetch(wfrag + (size_t)(((ng2 >> 5) * 32 + (ng2 & 31)) * 16) * 512 +
                             lane * 16, 0, 3);
    }
#pragma unroll 4
    for (int ks = 0; ks < 16; ++ks) {
      v16bf a  = *(const v16bf*)(ldsA + ks * 512 + lane * 16);
      v16bf bm = *(const v16bf*)(wb + ks * 512 + lane * 16);
      acc = wmma_bf16(a, bm, acc);
    }
    int n = nt * 16 + col;
    float bias = biasCat[m * 512 + n];   // branch-free
    int head = nt >> 2;
    int d0   = (nt & 3) * 16;
#pragma unroll
    for (int r = 0; r < 8; ++r) {
      float val = acc[r] + bias;
      int M = r + ((lane >> 4) << 3);       // t-row within tile
      if (m == 0) {
        // q as A-fragment: rows=t, K=d
        int d = d0 + col;
        int ksq = d >> 5, kk = d & 31;
        int hi = (kk >> 3) & 1;
        int lp = M | (hi << 4);
        int ep = (kk & 7) + ((kk >> 4) << 3);
        qf[(size_t)(((bp * 8 + head) * 16 + mt) * 2 + ksq) * 512 + lp * 16 + ep] =
            (__bf16)val;
      } else if (m == 1) {
        // k^T as B-fragment: K=d, col=s  (s-tile index == mt, col_s == M)
        int d = d0 + col;
        int ksk = d >> 5, kk = d & 31;
        int hi = (kk >> 3) & 1;
        int lp = M | (hi << 4);
        int ep = (kk & 7) + ((kk >> 4) << 3);
        kf[(size_t)(((bp * 8 + head) * 16 + mt) * 2 + ksk) * 512 + lp * 16 + ep] =
            (__bf16)val;
      } else {
        // v as B-fragment: K=s, col=d
        int s = mt * 16 + M;
        int ntd = nt & 3;
        int kss = s >> 5, kk = s & 31;
        int hi = (kk >> 3) & 1;
        int lp = col | (hi << 4);
        int ep = (kk & 7) + ((kk >> 4) << 3);
        vf[(size_t)(((bp * 8 + head) * 4 + ntd) * 8 + kss) * 512 + lp * 16 + ep] =
            (__bf16)val;
      }
    }
  }
}

// ---------------------------------------------------------------------------
// K2: flash-style causal attention. One wave per (bp, head, mt); online
// softmax over s-chunks of 32; P re-swizzled D->A layout via per-wave LDS.
// ---------------------------------------------------------------------------
__global__ void __launch_bounds__(256)
attn_kernel(const __bf16* __restrict__ qf, const __bf16* __restrict__ kf,
            const __bf16* __restrict__ vf, __bf16* __restrict__ af) {
  __shared__ __align__(32) __bf16 ldsP[8 * 32 * 16];   // 1 KB per wave
  int tid  = threadIdx.x;
  int w    = __builtin_amdgcn_readfirstlane(tid >> 5);
  int lane = tid & 31, col = lane & 15;
  int widx = blockIdx.x * 8 + w;        // 0..16383 (scalar)
  int bp   = widx >> 7;
  int head = (widx >> 4) & 7;
  int mt   = widx & 15;

  const __bf16* qb = qf + (size_t)((bp * 8 + head) * 16 + mt) * 1024;
  v16bf qa0 = *(const v16bf*)(qb + lane * 16);
  v16bf qa1 = *(const v16bf*)(qb + 512 + lane * 16);
  const __bf16* kb = kf + (size_t)(bp * 8 + head) * 16384;
  const __bf16* vb = vf + (size_t)(bp * 8 + head) * 16384;
  __bf16* lp = ldsP + w * 512;

  v8f o0 = {}, o1 = {}, o2 = {}, o3 = {};
  float mrun[8], lrun[8];
#pragma unroll
  for (int r = 0; r < 8; ++r) { mrun[r] = -1e30f; lrun[r] = 0.f; }

  int cmax = (mt * 16 + 15) >> 5;
  for (int c = 0; c <= cmax; ++c) {
    // prefetch next chunk's k^T fragments while this chunk computes
    if (c < cmax)
      __builtin_prefetch(kb + (size_t)(2 * (c + 1)) * 1024 + lane * 16, 0, 3);

    // S tiles: q(16x64) @ k^T(64x32) -> two 16x16 f32 tiles
    v8f s0 = {}, s1 = {};
    s0 = wmma_bf16(qa0, *(const v16bf*)(kb + (size_t)((2 * c) * 2 + 0) * 512 + lane * 16), s0);
    s0 = wmma_bf16(qa1, *(const v16bf*)(kb + (size_t)((2 * c) * 2 + 1) * 512 + lane * 16), s0);
    s1 = wmma_bf16(qa0, *(const v16bf*)(kb + (size_t)((2 * c + 1) * 2 + 0) * 512 + lane * 16), s1);
    s1 = wmma_bf16(qa1, *(const v16bf*)(kb + (size_t)((2 * c + 1) * 2 + 1) * 512 + lane * 16), s1);

    int sc0 = 32 * c + col, sc1 = sc0 + 16;
    float p0[8], p1[8];
#pragma unroll
    for (int r = 0; r < 8; ++r) {
      int trow = mt * 16 + r + ((lane >> 4) << 3);
      p0[r] = (sc0 <= trow) ? s0[r] * 0.125f : -1e30f;   // scale = KQ^-0.5
      p1[r] = (sc1 <= trow) ? s1[r] * 0.125f : -1e30f;
    }
#pragma unroll
    for (int r = 0; r < 8; ++r) {
      float mx = fmaxf(p0[r], p1[r]);
      for (int off = 1; off < 16; off <<= 1)
        mx = fmaxf(mx, __shfl_xor(mx, off, 32));         // 16-lane row groups
      float mnew = fmaxf(mrun[r], mx);
      float alpha = __expf(mrun[r] - mnew);
      mrun[r] = mnew;
      p0[r] = (p0[r] > -1e29f) ? __expf(p0[r] - mnew) : 0.f;
      p1[r] = (p1[r] > -1e29f) ? __expf(p1[r] - mnew) : 0.f;
      float rs = p0[r] + p1[r];
      for (int off = 1; off < 16; off <<= 1)
        rs += __shfl_xor(rs, off, 32);
      lrun[r] = lrun[r] * alpha + rs;
      o0[r] *= alpha; o1[r] *= alpha; o2[r] *= alpha; o3[r] *= alpha;
      // stash P into LDS in A-fragment layout (D-layout -> A-layout swizzle)
      int M  = r + ((lane >> 4) << 3);
      int hi = (col >> 3) & 1;
      int lpn = M | (hi << 4);
      int ep  = col & 7;
      lp[lpn * 16 + ep]     = (__bf16)p0[r];   // K = col
      lp[lpn * 16 + ep + 8] = (__bf16)p1[r];   // K = col + 16
    }
    asm volatile("s_wait_dscnt 0x0" ::: "memory");
    v16bf pa = *(const v16bf*)(lp + lane * 16);
    // O += P(16x32) @ V(32x64)
    o0 = wmma_bf16(pa, *(const v16bf*)(vb + (size_t)(0 * 8 + c) * 512 + lane * 16), o0);
    o1 = wmma_bf16(pa, *(const v16bf*)(vb + (size_t)(1 * 8 + c) * 512 + lane * 16), o1);
    o2 = wmma_bf16(pa, *(const v16bf*)(vb + (size_t)(2 * 8 + c) * 512 + lane * 16), o2);
    o3 = wmma_bf16(pa, *(const v16bf*)(vb + (size_t)(3 * 8 + c) * 512 + lane * 16), o3);
  }

  // normalize by softmax denominator and scatter into A-fragments for Wo GEMM
#pragma unroll
  for (int r = 0; r < 8; ++r) {
    float inv = 1.0f / lrun[r];
    int M = r + ((lane >> 4) << 3);
    float ov[4] = {o0[r] * inv, o1[r] * inv, o2[r] * inv, o3[r] * inv};
#pragma unroll
    for (int nt = 0; nt < 4; ++nt) {
      int d = nt * 16 + col;
      int h = head * 64 + d;
      int ksa = h >> 5, kk = h & 31;
      int hi = (kk >> 3) & 1;
      int lpn = M | (hi << 4);
      int ep  = (kk & 7) + ((kk >> 4) << 3);
      af[(size_t)((bp * 16 + mt) * 16 + ksa) * 512 + lpn * 16 + ep] = (__bf16)ov[nt];
    }
  }
}

// ---------------------------------------------------------------------------
// K3: output projection (WMMA) + bias + residual + LayerNorm, write fp32 out
// in the original (b,t,p,e) layout. Block = (bp, mt); 8 waves * 4 n-tiles.
// ---------------------------------------------------------------------------
__global__ void __launch_bounds__(256)
proj_ln_kernel(const float* __restrict__ x, const float* __restrict__ bo,
               const float* __restrict__ ln_g, const float* __restrict__ ln_b,
               const __bf16* __restrict__ wof, const __bf16* __restrict__ af,
               float* __restrict__ out) {
  __shared__ __align__(32) float ly[16 * 512];   // 32 KB residual+proj tile
  int bp = blockIdx.x >> 4;
  int mt = blockIdx.x & 15;
  int b = bp >> 6, p = bp & 63;
  int tid  = threadIdx.x;
  int w    = __builtin_amdgcn_readfirstlane(tid >> 5);
  int lane = tid & 31, col = lane & 15;

  const __bf16* ab = af + (size_t)(bp * 16 + mt) * 8192;
  for (int j = 0; j < 4; ++j) {
    int nt = w * 4 + j;                          // scalar
    v8f acc = {};
    const __bf16* wb = wof + (size_t)nt * 8192;
    if (j + 1 < 4)
      __builtin_prefetch(wof + (size_t)(nt + 1) * 8192 + lane * 16, 0, 3);
#pragma unroll 4
    for (int ks = 0; ks < 16; ++ks) {
      v16bf a  = *(const v16bf*)(ab + ks * 512 + lane * 16);
      v16bf bm = *(const v16bf*)(wb + ks * 512 + lane * 16);
      acc = wmma_bf16(a, bm, acc);
    }
    int n = nt * 16 + col;
    float bias = bo[n];
#pragma unroll
    for (int r = 0; r < 8; ++r) {
      int M = r + ((lane >> 4) << 3);
      int t = mt * 16 + M;
      float xv = x[(size_t)((b * 256 + t) * 64 + p) * 512 + n];
      ly[M * 512 + n] = acc[r] + bias + xv;
    }
  }
  __syncthreads();

  for (int rr = 0; rr < 2; ++rr) {
    int row = w * 2 + rr;
    int t = mt * 16 + row;
    float sum = 0.f;
#pragma unroll
    for (int i = 0; i < 16; ++i) sum += ly[row * 512 + lane + i * 32];
    for (int off = 1; off < 32; off <<= 1) sum += __shfl_xor(sum, off, 32);
    float mu = sum * (1.0f / 512.0f);
    float vs = 0.f;
#pragma unroll
    for (int i = 0; i < 16; ++i) {
      float d = ly[row * 512 + lane + i * 32] - mu;
      vs += d * d;
    }
    for (int off = 1; off < 32; off <<= 1) vs += __shfl_xor(vs, off, 32);
    float rstd = rsqrtf(vs * (1.0f / 512.0f) + 1e-5f);
    float* orow = out + (size_t)((b * 256 + t) * 64 + p) * 512;
#pragma unroll
    for (int i = 0; i < 16; ++i) {
      int e = lane + i * 32;
      orow[e] = (ly[row * 512 + e] - mu) * rstd * ln_g[e] + ln_b[e];
    }
  }
}

// ---------------------------------------------------------------------------
extern "C" void kernel_launch(void* const* d_in, const int* in_sizes, int n_in,
                              void* d_out, int out_size, void* d_ws,
                              size_t ws_size, hipStream_t stream) {
  const float* x    = (const float*)d_in[0];
  const float* Wq   = (const float*)d_in[1];
  const float* bq   = (const float*)d_in[2];
  const float* Wk   = (const float*)d_in[3];
  const float* bk   = (const float*)d_in[4];
  const float* Wv   = (const float*)d_in[5];
  const float* bv   = (const float*)d_in[6];
  const float* Wo   = (const float*)d_in[7];
  const float* bo   = (const float*)d_in[8];
  const float* ln_g = (const float*)d_in[9];
  const float* ln_b = (const float*)d_in[10];

  __bf16* ws    = (__bf16*)d_ws;
  __bf16* wfrag = ws;                       // 3*262144 (Wq,Wk,Wv) + 262144 (Wo)
  __bf16* wof   = ws + (size_t)786432;
  __bf16* qf    = ws + (size_t)1048576;
  __bf16* kf    = qf + (size_t)16777216;
  __bf16* vf    = kf + (size_t)16777216;
  __bf16* af    = vf + (size_t)16777216;
  float*  biasCat = (float*)(af + (size_t)16777216);   // 1536 floats

  pack_weights<<<4096, 256, 0, stream>>>(Wq, Wk, Wv, Wo, wfrag);
  pack_bias<<<6, 256, 0, stream>>>(bq, bk, bv, biasCat);
  qkv_kernel<<<2048, 256, 0, stream>>>(x, biasCat, wfrag, qf, kf, vf);
  attn_kernel<<<2048, 256, 0, stream>>>(qf, kf, vf, af);
  proj_ln_kernel<<<2048, 256, 0, stream>>>(x, bo, ln_g, ln_b, wof, af,
                                           (float*)d_out);
}
